// TreensformerBlockV4_50285477101583
// MI455X (gfx1250) — compile-verified, matrix operand order checked by reading, not run on hardware
//
#include <hip/hip_runtime.h>
#include <cstddef>
#include <cstdint>

using f16 = _Float16;
typedef __attribute__((ext_vector_type(16))) _Float16 v16h;
typedef __attribute__((ext_vector_type(8)))  _Float16 v8h;
typedef __attribute__((ext_vector_type(8)))  float    v8f;
typedef __attribute__((ext_vector_type(4)))  unsigned int u32x4;
typedef __attribute__((ext_vector_type(8)))  int i32x8;
typedef __attribute__((ext_vector_type(4)))  int i32x4;

#define DEVFN static __device__ __forceinline__

// ------------------------- WMMA helpers -------------------------

DEVFN v8f wmma_f16(v16h a, v16h b, v8f c) {
  return __builtin_amdgcn_wmma_f32_16x16x32_f16(
      /*neg_a=*/false, a, /*neg_b=*/false, b,
      /*c_mod=*/(short)0, c, /*reuse_a=*/false, /*reuse_b=*/false);
}

// A-layout fragment (16 rows x 32 K) from a row-major LDS tile.
// Lane l: row (l&15); K chunks at base 8*(l>>4) and base+16.
DEVFN v16h ldfrag(const f16* p, int stride) {
  int lane = threadIdx.x & 31;
  const f16* q = p + (lane & 15) * stride + ((lane >> 4) << 3);
  v8h lo = *(const v8h*)q;
  v8h hi = *(const v8h*)(q + 16);
  v16h f;
#pragma unroll
  for (int i = 0; i < 8; ++i) { f[i] = lo[i]; f[i + 8] = hi[i]; }
  return f;
}

// ------------------------- TDM (Tensor Data Mover) -------------------------
// 2D tile DMA: global (row-major, row_stride elems of f16) -> LDS with padding.
// pad_interval code: pad after 2^pi * 8 bytes; pad_amount code: (pa+1) DWORDs.
DEVFN void tdm_raw(u32x4 g0, i32x8 g1) {
#if __clang_major__ >= 23
  __builtin_amdgcn_tensor_load_to_lds(g0, g1, (i32x4)0, (i32x4)0, (i32x8)0, 0);
#else
  __builtin_amdgcn_tensor_load_to_lds(g0, g1, (i32x4)0, (i32x4)0, 0);
#endif
}

DEVFN void tdm_issue(const f16* gptr, unsigned int lds_off, int tile_k, int tile_rows,
                     int row_stride, int pad_int, int pad_amt) {
  unsigned long long ga = (unsigned long long)(uintptr_t)gptr;
  u32x4 g0;
  g0[0] = 1u;                         // count=1, no gather
  g0[1] = lds_off;                    // lds_addr (bytes)
  g0[2] = (unsigned int)ga;           // global_addr[31:0]
  g0[3] = (unsigned int)(ga >> 32) | (2u << 30);  // addr[56:32] | type=2
  i32x8 g1;
  g1[0] = (1 << 16) | (1 << 20) | (pad_int << 22) | (pad_amt << 25);  // 2B, pad_en
  g1[1] = 0;                          // atomic_barrier=0, tensor_dim0 lo16 = 0
  g1[2] = 0x4000;                     // tensor_dim0 = 1<<30 (hi16), dim1 lo16 = 0
  g1[3] = 0x4000 | (tile_k << 16);    // tensor_dim1 hi16, tile_dim0
  g1[4] = tile_rows;                  // tile_dim1 ; tile_dim2 = 0
  g1[5] = row_stride;                 // tensor_dim0_stride[31:0]
  g1[6] = 0;
  g1[7] = 0;
  tdm_raw(g0, g1);
}

DEVFN unsigned int lds_off_of(const void* p) {
  return (unsigned int)(uintptr_t)p;  // low 32 bits of generic LDS addr = LDS offset
}

DEVFN float gelu_f(float x) {
  return 0.5f * x * (1.0f + tanhf(0.7978845608028654f * (x + 0.044715f * x * x * x)));
}

DEVFN float redmax16(float v) {
#pragma unroll
  for (int m = 8; m >= 1; m >>= 1) v = fmaxf(v, __shfl_xor(v, m, 32));
  return v;
}
DEVFN float redsum16(float v) {
#pragma unroll
  for (int m = 8; m >= 1; m >>= 1) v += __shfl_xor(v, m, 32);
  return v;
}

// ------------------------- small kernels -------------------------

__global__ __launch_bounds__(256) void cvt_f32_to_f16(const float* __restrict__ src,
                                                      f16* __restrict__ dst, int n) {
  int i = blockIdx.x * 256 + threadIdx.x;
  if (i < n) dst[i] = (f16)src[i];
}

// wt[n][k] = w[k][n] (f32->f16), zero-pad rows n >= N up to Npad.
__global__ __launch_bounds__(256) void transpose_w_kernel(const float* __restrict__ w,
                                                          f16* __restrict__ wt, int K,
                                                          int N, int Npad) {
  size_t idx = (size_t)blockIdx.x * 256 + threadIdx.x;
  if (idx >= (size_t)Npad * K) return;
  int k = (int)(idx % K);
  int n = (int)(idx / K);
  wt[idx] = (n < N) ? (f16)w[(size_t)k * N + n] : (f16)0.f;
}

// LayerNorm over last dim R=192; one wave per row; 8 rows / 256-thr block.
__global__ __launch_bounds__(256) void ln_kernel(const float* __restrict__ x,
                                                 const float* __restrict__ g,
                                                 const float* __restrict__ b,
                                                 f16* __restrict__ out) {
  int row  = blockIdx.x * 8 + (threadIdx.x >> 5);
  int lane = threadIdx.x & 31;
  const float* xr = x + (size_t)row * 192;
  float v[6], s = 0.f, s2 = 0.f;
#pragma unroll
  for (int i = 0; i < 6; ++i) { v[i] = xr[lane + 32 * i]; s += v[i]; s2 += v[i] * v[i]; }
#pragma unroll
  for (int m = 16; m >= 1; m >>= 1) { s += __shfl_xor(s, m, 32); s2 += __shfl_xor(s2, m, 32); }
  float mu  = s * (1.f / 192.f);
  float var = s2 * (1.f / 192.f) - mu * mu;
  float inv = rsqrtf(var + 1e-5f);
  f16* orow = out + (size_t)row * 192;
#pragma unroll
  for (int i = 0; i < 6; ++i) {
    int c = lane + 32 * i;
    orow[c] = (f16)((v[i] - mu) * inv * g[c] + b[c]);
  }
}

__global__ __launch_bounds__(256) void equalize_kernel(float* __restrict__ x, int s, int lev) {
  int nb = 32 / s;
  int total = 8 * nb * nb * 192;
  int idx = blockIdx.x * 256 + threadIdx.x;
  if (idx >= total) return;
  int r = idx % 192; int t = idx / 192;
  int bx = t % nb; t /= nb; int by = t % nb; int b = t / nb;
  float sum = 0.f;
  for (int dy = 0; dy < s; ++dy)
    for (int dx = 0; dx < s; ++dx) {
      size_t off = ((((size_t)b * 32 + by * s + dy) * 32 + bx * s + dx) * 4 + lev) * 192 + r;
      sum += x[off];
    }
  float avg = sum / (float)(s * s);
  for (int dy = 0; dy < s; ++dy)
    for (int dx = 0; dx < s; ++dx) {
      size_t off = ((((size_t)b * 32 + by * s + dy) * 32 + bx * s + dx) * 4 + lev) * 192 + r;
      x[off] = avg;
    }
}

__global__ __launch_bounds__(192) void colavg_kernel(const f16* __restrict__ xt,
                                                     float* __restrict__ avg) {
  int b = blockIdx.x;
  int col = blockIdx.y * 192 + threadIdx.x;
  float s = 0.f;
  const f16* p = xt + (size_t)b * 1024 * 768 + col;
  for (int n = 0; n < 1024; ++n) s += (float)p[(size_t)n * 768];
  avg[b * 768 + col] = s * (1.f / 1024.f);
}

__global__ __launch_bounds__(256) void build_full_kernel(const f16* __restrict__ xt,
                                                         const float* __restrict__ avg,
                                                         f16* __restrict__ full, int iR) {
  size_t idx = (size_t)blockIdx.x * 256 + threadIdx.x;
  if (idx >= (size_t)8192 * 768) return;
  int c = (int)(idx % 768);
  int b = (int)((idx / 768) >> 10);
  full[idx] = (c < iR) ? (f16)avg[b * 768 + c] : xt[idx];
}

// ------------------------- GEMM (WMMA + TDM double-buffered) -------------------------
// C(M,N) = A(M,K) @ Bt(N,K)^T + bias ; A row-major f16, Bt = B^T row-major f16
// (N padded to 128 in Bt). 256 thr = 8 waves; 128x128 tile; wave tile 64x32.
// LDS rows padded to 56 elems (112B) by TDM: 64B row + 48B pad -> conflict-free b128.
__global__ __launch_bounds__(256) void gemm_f16_kernel(
    const f16* __restrict__ A, const f16* __restrict__ Bt,
    const float* __restrict__ bias, int M, int N, int K, int do_gelu,
    f16* __restrict__ outh, float* __restrict__ outf, const float* __restrict__ res,
    int col_off, int ldo) {
  __shared__ __align__(16) f16 sA[2][128 * 56];
  __shared__ __align__(16) f16 sB[2][128 * 56];
  int tid = threadIdx.x, lane = tid & 31, wave = tid >> 5;
  int wm = wave >> 2, wn = wave & 3;
  int m0 = blockIdx.y * 128, n0 = blockIdx.x * 128;
  bool issuer = (wave == 0);

  v8f acc[4][2];
#pragma unroll
  for (int i = 0; i < 4; ++i)
#pragma unroll
    for (int j = 0; j < 2; ++j) acc[i][j] = (v8f)0.f;

  const f16* Ab = A + (size_t)m0 * K;
  const f16* Bb = Bt + (size_t)n0 * K;
  int nk = K / 32;

  if (issuer) {
    tdm_issue(Ab, lds_off_of(&sA[0][0]), 32, 128, K, 3, 11);
    tdm_issue(Bb, lds_off_of(&sB[0][0]), 32, 128, K, 3, 11);
  }

  for (int t = 0; t < nk; ++t) {
    int cur = t & 1;
    if (issuer) {
      if (t + 1 < nk) {
        int k0 = (t + 1) * 32;
        tdm_issue(Ab + k0, lds_off_of(&sA[cur ^ 1][0]), 32, 128, K, 3, 11);
        tdm_issue(Bb + k0, lds_off_of(&sB[cur ^ 1][0]), 32, 128, K, 3, 11);
        __builtin_amdgcn_s_wait_tensorcnt(2);  // retire tiles for `cur`
      } else {
        __builtin_amdgcn_s_wait_tensorcnt(0);  // last pair fully landed
      }
    }
    __syncthreads();  // tiles for `cur` ready for all waves

    v16h af[4], bf[2];
#pragma unroll
    for (int i = 0; i < 4; ++i) af[i] = ldfrag(&sA[cur][(wm * 64 + i * 16) * 56], 56);
#pragma unroll
    for (int j = 0; j < 2; ++j) bf[j] = ldfrag(&sB[cur][(wn * 32 + j * 16) * 56], 56);
#pragma unroll
    for (int i = 0; i < 4; ++i)
#pragma unroll
      for (int j = 0; j < 2; ++j) acc[i][j] = wmma_f16(af[i], bf[j], acc[i][j]);

    __syncthreads();  // everyone done reading `cur` before it is refilled
  }

  // epilogue; C layout: lane l, elem r -> M = r + 8*(l>=16), N = l&15
  int l16 = lane & 15, hh = lane >> 4;
#pragma unroll
  for (int j = 0; j < 2; ++j) {
    int n = n0 + wn * 32 + j * 16 + l16;
    if (n >= N) continue;
    float bv = bias[n];
#pragma unroll
    for (int i = 0; i < 4; ++i) {
#pragma unroll
      for (int r = 0; r < 8; ++r) {
        int m = m0 + wm * 64 + i * 16 + r + hh * 8;
        float v = acc[i][j][r] + bv;
        if (do_gelu) v = gelu_f(v);
        size_t o = (size_t)m * ldo + col_off + n;
        if (outh) outh[o] = (f16)v;
        if (outf) outf[o] = (res ? res[o] : 0.f) + v;
      }
    }
  }
}

// ------------------------- Flash attention -------------------------
// qkv (f16): [8192, 2304]; q cols [0,768), k [768,1536), v [1536,2304)
// Grid (qblock, head, batch); 8 waves x 16 query rows = 128 rows / WG.
__global__ __launch_bounds__(256) void attn_kernel(const f16* __restrict__ qkv,
                                                   f16* __restrict__ y) {
  int qb = blockIdx.x, head = blockIdx.y, b = blockIdx.z;
  int tid = threadIdx.x, lane = tid & 31, wave = tid >> 5;
  int l16 = lane & 15, hh = lane >> 4;

  __shared__ __align__(16) f16 sK[32 * 72];     // [key][d], stride 72 (TDM pad)
  __shared__ __align__(16) f16 sVt[64 * 56];    // [d][key], stride 56 (manual)
  __shared__ __align__(16) f16 sP[8][16 * 40];  // per-wave P transpose

  const size_t ld = 2304;
  size_t rowbase = (size_t)b * 1024;
  int qrow0 = qb * 128 + wave * 16;

  v16h qf[2];
  {
    const f16* qp = qkv + (rowbase + qrow0 + l16) * ld + head * 64;
#pragma unroll
    for (int f = 0; f < 2; ++f) {
      int kb = f * 32 + hh * 8;
      v8h lo = *(const v8h*)(qp + kb);
      v8h hi = *(const v8h*)(qp + kb + 16);
#pragma unroll
      for (int i = 0; i < 8; ++i) { qf[f][i] = lo[i]; qf[f][i + 8] = hi[i]; }
    }
  }

  v8f acc[4];
#pragma unroll
  for (int t = 0; t < 4; ++t) acc[t] = (v8f)0.f;
  float rm[8], rl[8];
#pragma unroll
  for (int r = 0; r < 8; ++r) { rm[r] = -1e30f; rl[r] = 0.f; }

  for (int kb = 0; kb < 32; ++kb) {
    __syncthreads();  // previous tiles fully consumed
    // K tile via TDM: 32 keys x 64 d, rows 128B + 16B pad -> stride 72 elems
    if (wave == 0)
      tdm_issue(qkv + (rowbase + kb * 32) * ld + 768 + head * 64,
                lds_off_of(&sK[0]), 64, 32, 2304, 4, 3);
    // V^T manual transpose staging
    {
      int key = tid >> 3, dc = (tid & 7) * 8;
      const f16* vp = qkv + (rowbase + kb * 32 + key) * ld + 1536 + head * 64 + dc;
      v8h vv = *(const v8h*)vp;
#pragma unroll
      for (int j = 0; j < 8; ++j) sVt[(dc + j) * 56 + key] = vv[j];
    }
    if (wave == 0) __builtin_amdgcn_s_wait_tensorcnt(0);
    __syncthreads();

    // S(16x32) = Q @ K^T as two 16x16 tiles
    v8f s[2];
    s[0] = (v8f)0.f; s[1] = (v8f)0.f;
#pragma unroll
    for (int t = 0; t < 2; ++t)
#pragma unroll
      for (int f = 0; f < 2; ++f) {
        v16h kf = ldfrag(&sK[(t * 16) * 72 + f * 32], 72);
        s[t] = wmma_f16(qf[f], kf, s[t]);
      }
#pragma unroll
    for (int t = 0; t < 2; ++t)
#pragma unroll
      for (int r = 0; r < 8; ++r) s[t][r] *= 0.125f;

    float mnew[8], scl[8];
#pragma unroll
    for (int r = 0; r < 8; ++r) {
      float mx = redmax16(fmaxf(s[0][r], s[1][r]));
      mnew[r] = fmaxf(rm[r], mx);
      scl[r] = __expf(rm[r] - mnew[r]);
      rm[r] = mnew[r];
    }
#pragma unroll
    for (int t = 0; t < 2; ++t)
#pragma unroll
      for (int r = 0; r < 8; ++r) s[t][r] = __expf(s[t][r] - mnew[r]);
#pragma unroll
    for (int r = 0; r < 8; ++r)
      rl[r] = rl[r] * scl[r] + redsum16(s[0][r] + s[1][r]);
#pragma unroll
    for (int t = 0; t < 4; ++t)
#pragma unroll
      for (int r = 0; r < 8; ++r) acc[t][r] *= scl[r];

    // P (C-layout) -> per-wave LDS -> A-fragment layout
    f16* myP = &sP[wave][0];
#pragma unroll
    for (int t = 0; t < 2; ++t)
#pragma unroll
      for (int r = 0; r < 8; ++r)
        myP[(r + 8 * hh) * 40 + t * 16 + l16] = (f16)s[t][r];
    asm volatile("s_wait_dscnt 0" ::: "memory");
    v16h pf = ldfrag(myP, 40);

#pragma unroll
    for (int dt = 0; dt < 4; ++dt) {
      v16h vf = ldfrag(&sVt[(dt * 16) * 56], 56);
      acc[dt] = wmma_f16(pf, vf, acc[dt]);
    }
  }

#pragma unroll
  for (int dt = 0; dt < 4; ++dt)
#pragma unroll
    for (int r = 0; r < 8; ++r) {
      int m = qrow0 + r + 8 * hh;
      float o = acc[dt][r] / rl[r];
      y[(rowbase + m) * 768 + head * 64 + dt * 16 + l16] = (f16)o;
    }
}

// ------------------------- host -------------------------

extern "C" void kernel_launch(void* const* d_in, const int* in_sizes, int n_in,
                              void* d_out, int out_size, void* d_ws, size_t ws_size,
                              hipStream_t stream) {
  constexpr int M = 8192, C = 768, R = 192, L = 4;
  const float* x      = (const float*)d_in[0];
  const float* ln1_g  = (const float*)d_in[1];
  const float* ln1_b  = (const float*)d_in[2];
  const float* ln2_g  = (const float*)d_in[3];
  const float* ln2_b  = (const float*)d_in[4];
  const float* w_qkv  = (const float*)d_in[5];
  const float* b_qkv  = (const float*)d_in[6];
  const float* w_proj = (const float*)d_in[7];
  const float* b_proj = (const float*)d_in[8];
  const float* w1     = (const float*)d_in[9];
  const float* b1     = (const float*)d_in[10];
  const float* w2     = (const float*)d_in[11];
  const float* b2     = (const float*)d_in[12];
  float* out = (float*)d_out;

  char* ws = (char*)d_ws;
  size_t off = 0;
  auto take = [&](size_t bytes) -> void* {
    void* p = ws + off;
    off += (bytes + 255) & ~(size_t)255;
    return p;
  };
  f16*   h16     = (f16*)take((size_t)M * C * 2);
  f16*   wqkvT   = (f16*)take((size_t)2304 * C * 2);   // [N=2304][K=768]
  f16*   wprojT  = (f16*)take((size_t)C * C * 2);      // [768][768]
  f16*   w1T     = (f16*)take((size_t)3072 * C * 2);   // [3072][768]
  f16*   w2T     = (f16*)take((size_t)256 * 3072 * 2); // [192->256 pad][3072]
  f16*   qkv16   = (f16*)take((size_t)M * 2304 * 2);
  f16*   y16     = (f16*)take((size_t)M * C * 2);
  float* x_eq    = (float*)take((size_t)M * C * 4);
  f16*   xt16    = (f16*)take((size_t)M * C * 2);
  f16*   full16  = (f16*)take((size_t)M * C * 2);
  f16*   hid16   = (f16*)take((size_t)M * 3072 * 2);
  float* avgbuf  = (float*)take((size_t)8 * C * 4);
  (void)ws_size; (void)n_in; (void)in_sizes; (void)out_size;

  auto transpose = [&](const float* w, f16* wt, int K, int N, int Npad) {
    size_t n = (size_t)Npad * K;
    transpose_w_kernel<<<(unsigned)((n + 255) / 256), 256, 0, stream>>>(w, wt, K, N, Npad);
  };
  transpose(w_qkv,  wqkvT,  C, 2304, 2304);
  transpose(w_proj, wprojT, C, C, C);
  transpose(w1,     w1T,    C, 3072, 3072);
  transpose(w2,     w2T,    3072, R, 256);

  // LN1
  ln_kernel<<<(M * L) / 8, 256, 0, stream>>>(x, ln1_g, ln1_b, h16);

  // QKV GEMM
  gemm_f16_kernel<<<dim3(2304 / 128, M / 128), 256, 0, stream>>>(
      h16, wqkvT, b_qkv, M, 2304, C, 0, qkv16, nullptr, nullptr, 0, 2304);

  // Attention
  attn_kernel<<<dim3(8, 12, 8), 256, 0, stream>>>(qkv16, y16);

  // Proj GEMM + residual add -> x_eq (fp32)
  gemm_f16_kernel<<<dim3(C / 128, M / 128), 256, 0, stream>>>(
      y16, wprojT, b_proj, M, C, C, 0, nullptr, x_eq, x, 0, C);

  // equalize_parents levels 1..3 in place
  for (int lev = 1; lev < 4; ++lev) {
    int s = 1 << lev;
    int nb = 32 / s;
    int total = 8 * nb * nb * R;
    equalize_kernel<<<(total + 255) / 256, 256, 0, stream>>>(x_eq, s, lev);
  }

  // LN2 -> tree working buffer
  ln_kernel<<<(M * L) / 8, 256, 0, stream>>>(x_eq, ln2_g, ln2_b, xt16);

  // tree_mlp: 4 sequential levels
  for (int i = 0; i < L; ++i) {
    const f16* A = xt16;
    if (i > 0) {
      colavg_kernel<<<dim3(8, i), 192, 0, stream>>>(xt16, avgbuf);
      build_full_kernel<<<(M * C + 255) / 256, 256, 0, stream>>>(xt16, avgbuf, full16,
                                                                 i * R);
      A = full16;
    }
    gemm_f16_kernel<<<dim3(3072 / 128, M / 128), 256, 0, stream>>>(
        A, w1T, b1, M, 3072, C, 1, hid16, nullptr, nullptr, 0, 3072);
    gemm_f16_kernel<<<dim3(2, M / 128), 256, 0, stream>>>(
        hid16, w2T, b2, M, R, 3072, 0, xt16, out, x_eq, i * R, C);
  }
}
